// CustomLSTM_24361054503564
// MI455X (gfx1250) — matrix operand order repaired and checked by reading, not compiled, not run on hardware
//
#include <hip/hip_runtime.h>
#include <hip/hip_bf16.h>

// ---------------------------------------------------------------------------
// CDNA5 (gfx1250) LSTM: bf16 WMMA GEMMs + TDM (tensor_load_to_lds) staging.
//   B=64, S=512, I=1024, H=1024, 4H=4096
// Workspace layout (bytes):
//   [0)                xW  f32 [32768][4096]   512 MB
//   [+536870912)       Xb  bf16 [32768][1024]   64 MB
//   [+603979776)       WT  bf16 [4096][1024]     8 MB   (W transposed)
//   [+612368384)       UT  bf16 [4096][1024]     8 MB   (U transposed)
//   [+620756992)       H0  bf16 [64][1024]      128 KB  (ping)
//   [+620888064)       H1  bf16 [64][1024]      128 KB  (pong)
//   [+621019136)       C   f32  [64][1024]      256 KB
// ---------------------------------------------------------------------------

typedef __bf16 bf16_t;
typedef __attribute__((ext_vector_type(16))) __bf16 v16bf;
typedef __attribute__((ext_vector_type(4)))  __bf16 v4bf;
typedef __attribute__((ext_vector_type(8)))  float  v8f;
typedef __attribute__((ext_vector_type(4)))  unsigned u32x4;
typedef __attribute__((ext_vector_type(8)))  int      i32x8;
typedef __attribute__((ext_vector_type(4)))  int      i32x4;

#if __has_builtin(__builtin_amdgcn_tensor_load_to_lds) && \
    __has_builtin(__builtin_amdgcn_s_wait_tensorcnt)
#define HAVE_TDM 1
#else
#define HAVE_TDM 0
#endif

union ABfrag { unsigned u[8]; v16bf v; };

// A-fragment (16x32 bf16, M x K) per ISA 7.12.2 "16-bit A-Matrix 16x32":
// lane L (0..15): row M=L, K groups 0..7,16..23 ; lanes 16..31: K 8..15,24..31.
// Two contiguous 16B runs per lane -> 2x ds_load_b128.
__device__ __forceinline__ v16bf load_a_frag(const bf16_t* As, int row_base,
                                             int lane, int stride) {
  const bf16_t* p = As + (row_base + (lane & 15)) * stride;
  const int kh = (lane >> 4) * 8;
  ABfrag r;
#pragma unroll
  for (int v = 0; v < 4; ++v) {
    r.u[v]     = *(const unsigned*)(p + kh + 2 * v);        // K = kh+2v, +1
    r.u[4 + v] = *(const unsigned*)(p + kh + 16 + 2 * v);   // K = kh+16+2v, +1
  }
  return r.v;
}

// B-fragment (32x16 bf16, K x N) from a column-major LDS tile Bs[col][k]:
// lane L (0..15): col N=L, K=0..7,16..23 ; lanes 16..31: K=8..15,24..31.
__device__ __forceinline__ v16bf load_b_frag(const bf16_t* Bs, int col_base,
                                             int lane, int stride) {
  const bf16_t* p = Bs + (col_base + (lane & 15)) * stride + ((lane >> 4) << 3);
  ABfrag r;
  const uint4* q = (const uint4*)p;   // 16B aligned (stride*2 = 80B)
  uint4 x0 = q[0], x1 = q[2];
  r.u[0] = x0.x; r.u[1] = x0.y; r.u[2] = x0.z; r.u[3] = x0.w;
  r.u[4] = x1.x; r.u[5] = x1.y; r.u[6] = x1.z; r.u[7] = x1.w;
  return r.v;
}

__device__ __forceinline__ v8f wmma_bf16(v16bf a, v16bf b, v8f c) {
  return __builtin_amdgcn_wmma_f32_16x16x32_bf16(false, a, false, b,
                                                 (short)0, c, false, false);
}

__device__ __forceinline__ float sigmoidf_(float x) {
  return 1.0f / (1.0f + __expf(-x));
}

// ---------------------------------------------------------------------------
// Tensor Data Mover: build a D# (ISA cdna5 §8) and issue TENSOR_LOAD_TO_LDS.
// Up-to-3D tile: addr = ga + 2*(x + y*s0 + z*s1); LDS written contiguously
// with pad_enable: +4 DWORDs after every 16 DWORDs (64B) of payload, i.e.
// an x-line of 32 bf16 lands on an 80B (=40 bf16) LDS pitch, matching the
// fragment readers above.
// ---------------------------------------------------------------------------
#if HAVE_TDM
__device__ __forceinline__ uint32_t lds_off(const void* p) {
  return (uint32_t)(uintptr_t)p;   // low 32 bits of flat LDS address = offset
}

__device__ __forceinline__ void tdm_load(
    uint64_t ga, uint32_t ldsa,
    uint32_t td0, uint32_t td1, uint32_t td2,
    uint32_t tl0, uint32_t tl1, uint32_t tl2,
    uint64_t s0, uint64_t s1) {
  // Group 0: count=1 | lds_addr | global_addr[56:0] | type=2
  u32x4 g0 = { 1u, ldsa, (uint32_t)ga,
               (uint32_t)((ga >> 32) & 0x1FFFFFFu) | (2u << 30) };
  // Group 1: data_size=1(2B), pad_enable, pad_interval=3(16 DW), pad_amount=3(4 DW)
  i32x8 g1;
  g1[0] = (int)((1u << 16) | (1u << 20) | (3u << 22) | (3u << 25));
  g1[1] = (int)((td0 & 0xFFFFu) << 16);                                // dim0 lo
  g1[2] = (int)(((td0 >> 16) & 0xFFFFu) | ((td1 & 0xFFFFu) << 16));    // dim0 hi|dim1 lo
  g1[3] = (int)(((td1 >> 16) & 0xFFFFu) | ((tl0 & 0xFFFFu) << 16));    // dim1 hi|tile0
  g1[4] = (int)((tl1 & 0xFFFFu) | ((tl2 & 0xFFFFu) << 16));            // tile1|tile2
  g1[5] = (int)(uint32_t)s0;                                           // dim0_stride lo
  g1[6] = (int)(((uint32_t)(s0 >> 32) & 0xFFFFu) |
                (((uint32_t)s1 & 0xFFFFu) << 16));                     // s0 hi|s1 lo
  g1[7] = (int)(uint32_t)(s1 >> 16);                                   // s1[47:16]
  // Group 2: tensor_dim2 (3D); group 3 unused (all-zero -> 2-group form).
  i32x4 g2 = { (int)td2, 0, 0, 0 };
  i32x4 g3 = { 0, 0, 0, 0 };
#if defined(__clang_major__) && __clang_major__ >= 23
  i32x8 g4 = { 0, 0, 0, 0, 0, 0, 0, 0 };
  __builtin_amdgcn_tensor_load_to_lds(g0, g1, g2, g3, g4, 0);
#else
  __builtin_amdgcn_tensor_load_to_lds(g0, g1, g2, g3, 0);
#endif
}
#endif  // HAVE_TDM

// ---------------------------------------------------------------------------
// Precision casts / transposes
// ---------------------------------------------------------------------------
__global__ void cast_f32_bf16_kernel(const float* __restrict__ in,
                                     bf16_t* __restrict__ out, int n) {
  int i = (blockIdx.x * 256 + threadIdx.x) * 4;
  if (i + 3 < n) {
    float4 v = *(const float4*)(in + i);
    v4bf o;
    o[0] = (bf16_t)v.x; o[1] = (bf16_t)v.y;
    o[2] = (bf16_t)v.z; o[3] = (bf16_t)v.w;
    *(v4bf*)(out + i) = o;
  } else {
    for (int j = i; j < n; ++j) out[j] = (bf16_t)in[j];
  }
}

// in[R][C] (f32) -> out[C][R] (bf16). One-time cost, bandwidth-trivial.
__global__ void transpose_cast_kernel(const float* __restrict__ in,
                                      bf16_t* __restrict__ out, int R, int C) {
  int i = blockIdx.x * 256 + threadIdx.x;
  if (i >= R * C) return;
  int r = i / C, c = i - r * C;
  out[(size_t)c * R + r] = (bf16_t)in[i];
}

__global__ void init_state_kernel(bf16_t* __restrict__ h0,
                                  bf16_t* __restrict__ h1,
                                  float* __restrict__ c, int n) {
  int i = blockIdx.x * 256 + threadIdx.x;
  if (i < n) { h0[i] = (bf16_t)0.0f; h1[i] = (bf16_t)0.0f; c[i] = 0.0f; }
}

// ---------------------------------------------------------------------------
// GEMM1: xW[32768][4096] = Xb[32768][1024] @ W + bias   (WT is [4096][1024])
// 8 wave32; workgroup tile 128x128; wave tile 32x64 (2x4 WMMA accumulators).
// TDM double-buffered: wave 0 DMAs the next A/B tiles while all waves compute.
// ---------------------------------------------------------------------------
__global__ __launch_bounds__(256) void gemm_xw_kernel(
    const bf16_t* __restrict__ X, const bf16_t* __restrict__ WT,
    const float* __restrict__ bias, float* __restrict__ XW,
    int M, int N, int K) {
  const int tid  = threadIdx.x;
  const int lane = tid & 31;
  const int wave = tid >> 5;
  const int bm = blockIdx.y * 128;
  const int bn = blockIdx.x * 128;
  const int mw = (wave >> 1) * 32;
  const int nw = (wave & 1) * 64;

  v8f acc[2][4] = {};

#if HAVE_TDM
  __shared__ __align__(16) bf16_t As[2][128 * 40];
  __shared__ __align__(16) bf16_t Bs[2][128 * 40];
  const int NCH = K / 32;
  if (wave == 0) {   // preload chunk 0 (TDM ignores EXEC; one instr per tile)
    tdm_load((uint64_t)(uintptr_t)(X + (size_t)bm * K),  lds_off(As[0]),
             32, 128, 0, 32, 128, 0, (uint64_t)K, 0);
    tdm_load((uint64_t)(uintptr_t)(WT + (size_t)bn * K), lds_off(Bs[0]),
             32, 128, 0, 32, 128, 0, (uint64_t)K, 0);
  }
  for (int ch = 0; ch < NCH; ++ch) {
    const int cur = ch & 1;
    if (wave == 0) {
      if (ch + 1 < NCH) {
        const int kc = (ch + 1) * 32;
        tdm_load((uint64_t)(uintptr_t)(X + (size_t)bm * K + kc),
                 lds_off(As[cur ^ 1]), 32, 128, 0, 32, 128, 0, (uint64_t)K, 0);
        tdm_load((uint64_t)(uintptr_t)(WT + (size_t)bn * K + kc),
                 lds_off(Bs[cur ^ 1]), 32, 128, 0, 32, 128, 0, (uint64_t)K, 0);
        __builtin_amdgcn_s_wait_tensorcnt(2);   // chunk `ch` pair retired
      } else {
        __builtin_amdgcn_s_wait_tensorcnt(0);
      }
    }
    __syncthreads();
    // Hoist all fragment loads, then burst the 8 WMMAs.
    v16bf a0 = load_a_frag(As[cur], mw,      lane, 40);
    v16bf a1 = load_a_frag(As[cur], mw + 16, lane, 40);
    v16bf b0 = load_b_frag(Bs[cur], nw,      lane, 40);
    v16bf b1 = load_b_frag(Bs[cur], nw + 16, lane, 40);
    v16bf b2 = load_b_frag(Bs[cur], nw + 32, lane, 40);
    v16bf b3 = load_b_frag(Bs[cur], nw + 48, lane, 40);
    acc[0][0] = wmma_bf16(a0, b0, acc[0][0]);
    acc[1][0] = wmma_bf16(a1, b0, acc[1][0]);
    acc[0][1] = wmma_bf16(a0, b1, acc[0][1]);
    acc[1][1] = wmma_bf16(a1, b1, acc[1][1]);
    acc[0][2] = wmma_bf16(a0, b2, acc[0][2]);
    acc[1][2] = wmma_bf16(a1, b2, acc[1][2]);
    acc[0][3] = wmma_bf16(a0, b3, acc[0][3]);
    acc[1][3] = wmma_bf16(a1, b3, acc[1][3]);
    __syncthreads();   // readers done before TDM refills this buffer
  }
#else
  __shared__ __align__(16) bf16_t As[1][128 * 40];
  __shared__ __align__(16) bf16_t Bs[1][128 * 40];
  for (int kc = 0; kc < K; kc += 32) {
    if (kc + 32 < K) {
      __builtin_prefetch(X  + (size_t)(bm + (tid >> 1)) * K + kc + 32, 0, 1);
      __builtin_prefetch(WT + (size_t)(bn + (tid >> 1)) * K + kc + 32, 0, 1);
    }
#pragma unroll
    for (int i = 0; i < 2; ++i) {
      int c = tid + i * 256, r = c >> 2, ko = (c & 3) * 8;
      *(uint4*)(As[0] + r * 40 + ko) =
          *(const uint4*)(X + (size_t)(bm + r) * K + kc + ko);
      *(uint4*)(Bs[0] + r * 40 + ko) =
          *(const uint4*)(WT + (size_t)(bn + r) * K + kc + ko);
    }
    __syncthreads();
    v16bf a0 = load_a_frag(As[0], mw,      lane, 40);
    v16bf a1 = load_a_frag(As[0], mw + 16, lane, 40);
#pragma unroll
    for (int nt = 0; nt < 4; ++nt) {
      v16bf b = load_b_frag(Bs[0], nw + nt * 16, lane, 40);
      acc[0][nt] = wmma_bf16(a0, b, acc[0][nt]);
      acc[1][nt] = wmma_bf16(a1, b, acc[1][nt]);
    }
    __syncthreads();
  }
#endif

  // Epilogue: C/D layout: lane col = lane&15, row = vgpr + (lane>>4)*8.
  const int ml = (lane >> 4) * 8;
  const int nl = lane & 15;
#pragma unroll
  for (int mt = 0; mt < 2; ++mt)
#pragma unroll
    for (int nt = 0; nt < 4; ++nt) {
      int col = bn + nw + nt * 16 + nl;
      float bv = bias[col];
#pragma unroll
      for (int v = 0; v < 8; ++v) {
        int row = bm + mw + mt * 16 + ml + v;
        XW[(size_t)row * N + col] = acc[mt][nt][v] + bv;
      }
    }
}

// ---------------------------------------------------------------------------
// Fused recurrent step t (grid = 32 WGs, one per 32 hidden units):
//   gates[64][4x32] = Hin@U-slice + xW[:,t,:]-slice, then cell update.
// TDM double-buffered: per K-chunk one 2D load for the h tile and one 3D load
// gathering the 4 gate strips of UT (x=32 k, y=32 cols stride H, z=4 gates
// stride H*H) into padded LDS. The scan is the latency-critical path, so the
// next chunk's DMA overlaps the current chunk's WMMAs.
// ---------------------------------------------------------------------------
__global__ __launch_bounds__(256) void lstm_step_kernel(
    const float* __restrict__ XW, const bf16_t* __restrict__ UT,
    const bf16_t* __restrict__ Hin, bf16_t* __restrict__ Hout,
    float* __restrict__ C, float* __restrict__ Hseq,
    float* __restrict__ HT, float* __restrict__ CT,
    int t, int S, int H) {
  __shared__ __align__(16) float gates[64 * 132];

  const int tid  = threadIdx.x;
  const int lane = tid & 31;
  const int wave = tid >> 5;
  const int nh0  = blockIdx.x * 32;
  const int mt   = (wave >> 1) * 16;
  const int gsel = (wave & 1) * 64;
  const int G4   = 4 * H;

  v8f acc[4] = {};

#if HAVE_TDM
  __shared__ __align__(16) bf16_t Hs[2][64 * 40];
  __shared__ __align__(16) bf16_t Us[2][128 * 40];
  const int NCH = H / 32;
  if (wave == 0) {
    tdm_load((uint64_t)(uintptr_t)Hin, lds_off(Hs[0]),
             32, 64, 0, 32, 64, 0, (uint64_t)H, 0);
    tdm_load((uint64_t)(uintptr_t)(UT + (size_t)nh0 * H), lds_off(Us[0]),
             32, 32, 4, 32, 32, 4, (uint64_t)H, (uint64_t)H * H);
  }
  for (int ch = 0; ch < NCH; ++ch) {
    const int cur = ch & 1;
    if (wave == 0) {
      if (ch + 1 < NCH) {
        const int kc = (ch + 1) * 32;
        tdm_load((uint64_t)(uintptr_t)(Hin + kc), lds_off(Hs[cur ^ 1]),
                 32, 64, 0, 32, 64, 0, (uint64_t)H, 0);
        tdm_load((uint64_t)(uintptr_t)(UT + (size_t)nh0 * H + kc),
                 lds_off(Us[cur ^ 1]),
                 32, 32, 4, 32, 32, 4, (uint64_t)H, (uint64_t)H * H);
        __builtin_amdgcn_s_wait_tensorcnt(2);
      } else {
        __builtin_amdgcn_s_wait_tensorcnt(0);
      }
    }
    __syncthreads();
    v16bf a  = load_a_frag(Hs[cur], mt, lane, 40);
    v16bf b0 = load_b_frag(Us[cur], gsel,      lane, 40);
    v16bf b1 = load_b_frag(Us[cur], gsel + 16, lane, 40);
    v16bf b2 = load_b_frag(Us[cur], gsel + 32, lane, 40);
    v16bf b3 = load_b_frag(Us[cur], gsel + 48, lane, 40);
    acc[0] = wmma_bf16(a, b0, acc[0]);
    acc[1] = wmma_bf16(a, b1, acc[1]);
    acc[2] = wmma_bf16(a, b2, acc[2]);
    acc[3] = wmma_bf16(a, b3, acc[3]);
    __syncthreads();
  }
#else
  __shared__ __align__(16) bf16_t Hs[1][64 * 40];
  __shared__ __align__(16) bf16_t Us[1][128 * 40];
  for (int kc = 0; kc < H; kc += 32) {
    {
      int r = tid >> 2, ko = (tid & 3) * 8;
      *(uint4*)(Hs[0] + r * 40 + ko) =
          *(const uint4*)(Hin + (size_t)r * H + kc + ko);
    }
#pragma unroll
    for (int i = 0; i < 2; ++i) {
      int c = tid + i * 256, lc = c >> 2, ko = (c & 3) * 8;
      int n = (lc >> 5) * H + nh0 + (lc & 31);
      *(uint4*)(Us[0] + lc * 40 + ko) =
          *(const uint4*)(UT + (size_t)n * H + kc + ko);
    }
    __syncthreads();
    v16bf a = load_a_frag(Hs[0], mt, lane, 40);
#pragma unroll
    for (int q = 0; q < 4; ++q) {
      v16bf b = load_b_frag(Us[0], gsel + q * 16, lane, 40);
      acc[q] = wmma_bf16(a, b, acc[q]);
    }
    __syncthreads();
  }
#endif

  // Add xW contribution, stash pre-activations in LDS.
  const int ml = (lane >> 4) * 8;
  const int nl = lane & 15;
#pragma unroll
  for (int q = 0; q < 4; ++q) {
    int lc = gsel + q * 16 + nl;
    int g  = lc >> 5, j = lc & 31;
    int n  = g * H + nh0 + j;
#pragma unroll
    for (int v = 0; v < 8; ++v) {
      int m = mt + ml + v;
      gates[m * 132 + lc] = acc[q][v] + XW[((size_t)m * S + t) * G4 + n];
    }
  }
  __syncthreads();

  // Elementwise LSTM cell update: 2048 units, 8 per thread.
#pragma unroll
  for (int i = 0; i < 8; ++i) {
    int p = tid * 8 + i;
    int m = p >> 5, j = p & 31;
    float ig = gates[m * 132 + j];
    float fg = gates[m * 132 + 32 + j];
    float gg = gates[m * 132 + 64 + j];
    float og = gates[m * 132 + 96 + j];
    float si = sigmoidf_(ig);
    float sf = sigmoidf_(fg);
    float sg = tanhf(gg);
    float so = sigmoidf_(og);
    size_t ci = (size_t)m * H + nh0 + j;
    float cn = sf * C[ci] + si * sg;
    float hn = so * tanhf(cn);
    C[ci] = cn;
    Hseq[((size_t)m * S + t) * H + nh0 + j] = hn;
    Hout[ci] = (bf16_t)hn;
    if (t == S - 1) { HT[ci] = hn; CT[ci] = cn; }
  }
}

// ---------------------------------------------------------------------------
extern "C" void kernel_launch(void* const* d_in, const int* in_sizes, int n_in,
                              void* d_out, int out_size, void* d_ws,
                              size_t ws_size, hipStream_t stream) {
  (void)in_sizes; (void)n_in; (void)out_size; (void)ws_size;
  constexpr int B = 64, S = 512, I = 1024, H = 1024;
  constexpr int M = B * S, N4 = 4 * H;

  const float* x    = (const float*)d_in[0];
  const float* W    = (const float*)d_in[1];
  const float* U    = (const float*)d_in[2];
  const float* bias = (const float*)d_in[3];

  char* ws = (char*)d_ws;
  float*  XW = (float*)ws;
  bf16_t* Xb = (bf16_t*)(ws + (size_t)M * N4 * 4);
  bf16_t* WT = (bf16_t*)((char*)Xb + (size_t)M * I * 2);
  bf16_t* UT = (bf16_t*)((char*)WT + (size_t)I * N4 * 2);
  bf16_t* H0 = (bf16_t*)((char*)UT + (size_t)I * N4 * 2);
  bf16_t* H1 = (bf16_t*)((char*)H0 + (size_t)B * H * 2);
  float*  C  = (float*)((char*)H1 + (size_t)B * H * 2);

  float* out = (float*)d_out;
  float* HT  = out + (size_t)B * S * H;
  float* CT  = HT + (size_t)B * H;

  {
    int n = M * I;
    cast_f32_bf16_kernel<<<(n / 4 + 255) / 256, 256, 0, stream>>>(x, Xb, n);
  }
  {
    int n = I * N4;
    transpose_cast_kernel<<<(n + 255) / 256, 256, 0, stream>>>(W, WT, I, N4);
    transpose_cast_kernel<<<(n + 255) / 256, 256, 0, stream>>>(U, UT, I, N4);
  }
  {
    int n = B * H;
    init_state_kernel<<<(n + 255) / 256, 256, 0, stream>>>(H0, H1, C, n);
  }
  {
    dim3 grid(N4 / 128, M / 128);   // (32, 256)
    gemm_xw_kernel<<<grid, 256, 0, stream>>>(Xb, WT, bias, XW, M, N4, I);
  }
  for (int t = 0; t < S; ++t) {
    bf16_t* Hin  = (t & 1) ? H1 : H0;
    bf16_t* Hout = (t & 1) ? H0 : H1;
    lstm_step_kernel<<<H / 32, 256, 0, stream>>>(XW, UT, Hin, Hout, C, out,
                                                 HT, CT, t, S, H);
  }
}
